// LSTM_25091198943855
// MI455X (gfx1250) — matrix operand order repaired and checked by reading, not compiled, not run on hardware
//
#include <hip/hip_runtime.h>
#include <hip/hip_bf16.h>

#define SEQ   1024
#define BATCH 256
#define NINP  128
#define NHID  256

typedef __attribute__((ext_vector_type(16))) __bf16 v16bf;
typedef __attribute__((ext_vector_type(8)))  __bf16 v8bf;
typedef __attribute__((ext_vector_type(8)))  float  v8f;

__device__ __forceinline__ float sigm_f(float x) {
    return 1.0f / (1.0f + __expf(-x));
}
__device__ __forceinline__ float tanh_f(float x) {
    // tanh(x) = 1 - 2/(1+exp(2x)); saturates correctly for large |x|
    return 1.0f - 2.0f / (1.0f + __expf(2.0f * x));
}

// ---------------------------------------------------------------------------
// Prep: bf16 weight copies, zero h double-buffer, reset per-group counters.
// ---------------------------------------------------------------------------
__global__ void lstm_prep_kernel(const float* __restrict__ W_ih,
                                 const float* __restrict__ W_hh,
                                 __bf16* __restrict__ WihB,
                                 __bf16* __restrict__ WhhB,
                                 __bf16* __restrict__ hbuf,
                                 int* __restrict__ cnt) {
    size_t tid    = (size_t)blockIdx.x * blockDim.x + threadIdx.x;
    size_t stride = (size_t)gridDim.x * blockDim.x;
    for (size_t i = tid; i < 512; i += stride)          // 16 counters, 128B apart
        cnt[i] = 0;
    for (size_t i = tid; i < (size_t)4 * NHID * NINP; i += stride)
        WihB[i] = (__bf16)W_ih[i];
    for (size_t i = tid; i < (size_t)4 * NHID * NHID; i += stride)
        WhhB[i] = (__bf16)W_hh[i];
    for (size_t i = tid; i < (size_t)2 * BATCH * NHID; i += stride)
        hbuf[i] = (__bf16)0.0f;
}

// ---------------------------------------------------------------------------
// Persistent fused LSTM. 32 blocks x 256 threads (8 waves -> 2 waves/SIMD,
// which fits exactly: each wave allocates 512 VGPRs, SIMD file >= 1024).
// Wave w = blk*8+wv owns batch-tile m=w>>4, hidden-tile n=w&15 (16x16 of h).
// The recurrence is block-diagonal over batch: h rows of batch-tile m are
// produced AND consumed only by blocks 2m..2m+1, so each pair of blocks
// syncs on its own counter (16 independent pipelines, 2-way barriers).
// B-fragments are loop-invariant -> compiler keeps them register-resident.
// c-state lives in registers for all 1024 steps; h double-buffered in global.
// ---------------------------------------------------------------------------
__launch_bounds__(256)
__global__ void lstm_main_kernel(const float* __restrict__ input,
                                 const float* __restrict__ b_ih,
                                 const float* __restrict__ b_hh,
                                 const __bf16* __restrict__ WihB,
                                 const __bf16* __restrict__ WhhB,
                                 __bf16* __restrict__ hbuf,
                                 int* __restrict__ cnt,
                                 float* __restrict__ out) {
    __shared__ __align__(16) __bf16 xs[2][16 * NINP];  // x stage, double-buffered
    __shared__ __align__(16) __bf16 hs[16 * NHID];     // h_{t-1} rows of this m-tile

    const int tid    = threadIdx.x;
    const int wv     = tid >> 5;
    const int lane   = tid & 31;
    const int w      = blockIdx.x * 8 + wv;   // 0..255
    const int mt     = w >> 4;                // batch tile
    const int nt     = w & 15;                // hidden tile
    const int col    = lane & 15;
    const int rbase  = (lane < 16) ? 0 : 8;   // C/D row group of this lane
    const int koff8  = (lane < 16) ? 0 : 8;   // unused; A-fragment also uses koff16
    const int koff16 = (lane < 16) ? 0 : 16;  // B-fragment K sub-offset

    int* my_cnt = cnt + mt * 32;              // per-m-group counter (128B stride)

    // Per-gate bias (depends on output column only); constant over time.
    float bias[4];
#pragma unroll
    for (int g = 0; g < 4; g++) {
        int c = g * NHID + nt * 16 + col;
        bias[g] = b_ih[c] + b_hh[c];
    }

    // Per-lane B-fragment base pointers (W row = output column of z).
    const __bf16* bih_base = WihB + (size_t)(nt * 16 + col) * NINP + koff16;
    const __bf16* bhh_base = WhhB + (size_t)(nt * 16 + col) * NHID + koff16;

    // Staging roles (256 threads, 16 rows).
    const int srow = tid >> 4;           // 0..15
    const int xc0  = (tid & 15) * 8;     // x: 8 contiguous f32 cols
    const int hc0  = (tid & 15) * 16;    // h: 16 contiguous bf16 cols

    v8f cstate; v8f hstate;
#pragma unroll
    for (int i = 0; i < 8; i++) { cstate[i] = 0.0f; hstate[i] = 0.0f; }

    // ---- pre-stage x[0] into xs[0] ----
    {
        const float* src = input + ((size_t)mt * 16 + srow) * NINP + xc0;
        __bf16* dst = &xs[0][srow * NINP + xc0];
#pragma unroll
        for (int j = 0; j < 8; j += 4) {
            float4 v = *(const float4*)(src + j);
            dst[j + 0] = (__bf16)v.x; dst[j + 1] = (__bf16)v.y;
            dst[j + 2] = (__bf16)v.z; dst[j + 3] = (__bf16)v.w;
        }
    }

    for (int t = 0; t < SEQ; t++) {
        const __bf16* hsrc = hbuf + (size_t)(t & 1) * BATCH * NHID;
        __bf16*       hdst = hbuf + (size_t)((t + 1) & 1) * BATCH * NHID;

        // ---- stage h_{t-1} rows of this m-tile into LDS ----
        {
            const uint4* src = (const uint4*)(hsrc + ((size_t)mt * 16 + srow) * NHID + hc0);
            uint4* dst = (uint4*)(hs + srow * NHID + hc0);
            dst[0] = src[0];
            dst[1] = src[1];
        }

        // ---- prefetch x[t+1] (global loads only; latency hidden under WMMAs) ----
        float4 xpre[2];
        if (t + 1 < SEQ) {
            const float* src = input + ((size_t)(t + 1) * BATCH + mt * 16 + srow) * NINP + xc0;
            xpre[0] = *(const float4*)(src + 0);
            xpre[1] = *(const float4*)(src + 4);
        }

        __syncthreads();   // hs ready; xs[t&1] (stored last iter) ready

        // ---- accumulators = bias ----
        v8f acc[4];
#pragma unroll
        for (int g = 0; g < 4; g++)
#pragma unroll
            for (int i = 0; i < 8; i++) acc[g][i] = bias[g];

        // ---- 12 k-tiles: kt 0..3 from x (K=128), kt 4..11 from h (K=256).
        //      A-fragment software-pipelined one k-tile ahead. ----
        const __bf16* xbase = &xs[t & 1][(lane & 15) * NINP + koff16];
        const __bf16* hbase = &hs[(lane & 15) * NHID + koff16];

        auto load_afrag = [&](int kt) -> v16bf {
            const __bf16* ap = (kt < 4) ? (xbase + kt * 32) : (hbase + (kt - 4) * 32);
            v8bf lo = *(const v8bf*)ap;
            v8bf hi = *(const v8bf*)(ap + 8);
            return __builtin_shufflevector(lo, hi,
                0, 1, 2, 3, 4, 5, 6, 7, 8, 9, 10, 11, 12, 13, 14, 15);
        };

        v16bf a_cur = load_afrag(0);
#pragma unroll
        for (int kt = 0; kt < 12; kt++) {
            v16bf a_nxt = (kt < 11) ? load_afrag(kt + 1) : a_cur;
#pragma unroll
            for (int g = 0; g < 4; g++) {
                const __bf16* bp = (kt < 4)
                    ? (bih_base + (size_t)g * NHID * NINP + kt * 32)
                    : (bhh_base + (size_t)g * NHID * NHID + (kt - 4) * 32);
                v16bf bfrag = *(const v16bf*)bp;   // loop-invariant -> stays in VGPRs
                acc[g] = __builtin_amdgcn_wmma_f32_16x16x32_bf16(
                    false, a_cur, false, bfrag, (short)0, acc[g], false, false);
            }
            a_cur = a_nxt;
        }

        // ---- gates + state update (c in registers) ----
#pragma unroll
        for (int i = 0; i < 8; i++) {
            float ig = sigm_f(acc[0][i]);
            float fg = sigm_f(acc[1][i]);
            float gg = tanh_f(acc[2][i]);
            float og = sigm_f(acc[3][i]);
            float c  = fg * cstate[i] + ig * gg;
            cstate[i] = c;
            hstate[i] = og * tanh_f(c);
        }

        // ---- publish h tile (bf16, double buffer) ----
        {
            __bf16* hp = hdst + (size_t)(mt * 16 + rbase) * NHID + nt * 16 + col;
#pragma unroll
            for (int r = 0; r < 8; r++) hp[(size_t)r * NHID] = (__bf16)hstate[r];
        }

        // ---- finish x[t+1] stage into the other LDS buffer ----
        if (t + 1 < SEQ) {
            __bf16* dst = &xs[(t + 1) & 1][srow * NINP + xc0];
            dst[0] = (__bf16)xpre[0].x; dst[1] = (__bf16)xpre[0].y;
            dst[2] = (__bf16)xpre[0].z; dst[3] = (__bf16)xpre[0].w;
            dst[4] = (__bf16)xpre[1].x; dst[5] = (__bf16)xpre[1].y;
            dst[6] = (__bf16)xpre[1].z; dst[7] = (__bf16)xpre[1].w;
        }

        // ---- 2-block barrier for this m-group (monotonic counter) ----
        __builtin_amdgcn_fence(__ATOMIC_RELEASE, "agent");
        __syncthreads();
        if (tid == 0) {
            __hip_atomic_fetch_add(my_cnt, 1, __ATOMIC_RELEASE, __HIP_MEMORY_SCOPE_AGENT);
            int target = 2 * (t + 1);
            while (__hip_atomic_load(my_cnt, __ATOMIC_ACQUIRE, __HIP_MEMORY_SCOPE_AGENT) < target)
                __builtin_amdgcn_s_sleep(1);
        }
        __syncthreads();
        __builtin_amdgcn_fence(__ATOMIC_ACQUIRE, "agent");
    }

    // ---- final h (fp32) -> d_out [BATCH, NHID] ----
    float* op = out + (size_t)(mt * 16 + rbase) * NHID + nt * 16 + col;
#pragma unroll
    for (int r = 0; r < 8; r++) op[(size_t)r * NHID] = hstate[r];
}

// ---------------------------------------------------------------------------
extern "C" void kernel_launch(void* const* d_in, const int* in_sizes, int n_in,
                              void* d_out, int out_size, void* d_ws, size_t ws_size,
                              hipStream_t stream) {
    const float* input = (const float*)d_in[0];
    const float* W_ih  = (const float*)d_in[1];
    const float* W_hh  = (const float*)d_in[2];
    const float* b_ih  = (const float*)d_in[3];
    const float* b_hh  = (const float*)d_in[4];

    char* ws = (char*)d_ws;
    int*    cnt  = (int*)ws;                                   // [0, 2048): 16 padded counters
    __bf16* WihB = (__bf16*)(ws + 2048);                       // 4H*NINP bf16 = 256 KB
    __bf16* WhhB = (__bf16*)(ws + 2048 + 262144);              // 4H*NHID bf16 = 512 KB
    __bf16* hbuf = (__bf16*)(ws + 2048 + 262144 + 524288);     // 2*B*H bf16  = 256 KB

    hipLaunchKernelGGL(lstm_prep_kernel, dim3(256), dim3(256), 0, stream,
                       W_ih, W_hh, WihB, WhhB, hbuf, cnt);
    hipLaunchKernelGGL(lstm_main_kernel, dim3(32), dim3(256), 0, stream,
                       input, b_ih, b_hh, WihB, WhhB, hbuf, cnt, (float*)d_out);
}
